// SparseMultiTokenAttention_25606595018754
// MI455X (gfx1250) — matrix operand order repaired and checked by reading, not compiled, not run on hardware
//
#include <hip/hip_runtime.h>
#include <hip/hip_bf16.h>

typedef __attribute__((ext_vector_type(16))) _Float16 v16h;
typedef __attribute__((ext_vector_type(8)))  _Float16 v8h;
typedef __attribute__((ext_vector_type(8)))  float    v8f;

#define DIM 768
#define HEADS 12
#define DH 64
#define NTOK 1024
#define BATCH 4
#define ROWS (BATCH * NTOK)          // 4096
#define QKV_N (3 * DIM)              // 2304
#define KSPARSE 512

// ---------------- fragment loaders ----------------
// A-matrix 16x32 f16 layout (ISA 7.12.2): lane L -> row (L&15);
// halves 0..7 = K kg..kg+7, halves 8..15 = K 16+kg..23+kg, kg = (L>>4)*8.
__device__ __forceinline__ v16h ld_fragA(const _Float16* p0, int ld, int lane) {
    const int r  = lane & 15;
    const int kg = (lane >> 4) << 3;
    const _Float16* p = p0 + (size_t)r * ld + kg;
    v8h lo = *(const v8h*)(p);
    v8h hi = *(const v8h*)(p + 16);
    v16h f;
#pragma unroll
    for (int i = 0; i < 8; ++i) { f[i] = lo[i]; f[i + 8] = hi[i]; }
    return f;
}

// B-matrix 32x16 f16 layout: lane L -> column (L&15);
// halves 0..7 = K koff..koff+7, halves 8..15 = K 16+koff..23+koff, koff = (L>>4)*8.
// Memory layout required: [n][k] row-major (i.e. B transposed), leading dim ld.
__device__ __forceinline__ v16h ld_fragB(const _Float16* p0, int ld, int lane) {
    const int n    = lane & 15;
    const int koff = (lane >> 4) << 3;
    const _Float16* p = p0 + (size_t)n * ld + koff;
    v8h lo = *(const v8h*)(p);
    v8h hi = *(const v8h*)(p + 16);
    v16h f;
#pragma unroll
    for (int i = 0; i < 8; ++i) { f[i] = lo[i]; f[i + 8] = hi[i]; }
    return f;
}

__device__ __forceinline__ v8f wmma16(v16h a, v16h b, v8f c) {
    return __builtin_amdgcn_wmma_f32_16x16x32_f16(false, a, false, b, (short)0, c, false, false);
}

__device__ __forceinline__ unsigned f2key(float f) {
    unsigned u = __float_as_uint(f);
    return (u & 0x80000000u) ? ~u : (u | 0x80000000u);
}

// per-lane prefetch (lowers to global_prefetch_b8)
__device__ __forceinline__ void pf(const _Float16* p) {
    __builtin_prefetch((const void*)p, 0, 3);
}

// ---------------- kernel A: precision/layout prep ----------------
__global__ void prep_kernel(const float* __restrict__ x,
                            const float* __restrict__ qkv_w,
                            const float* __restrict__ proj_w,
                            _Float16* __restrict__ x_h,
                            _Float16* __restrict__ wqkvT,
                            _Float16* __restrict__ wprojT) {
    const size_t NX = (size_t)ROWS * DIM;
    const size_t NQ = (size_t)DIM * QKV_N;
    const size_t NP = (size_t)DIM * DIM;
    size_t tid = (size_t)blockIdx.x * blockDim.x + threadIdx.x;
    if (tid < NX) {
        x_h[tid] = (_Float16)x[tid];
    } else if (tid < NX + NQ) {
        size_t t = tid - NX;                 // t = o*DIM + i
        int o = (int)(t / DIM), i = (int)(t % DIM);
        wqkvT[t] = (_Float16)qkv_w[(size_t)i * QKV_N + o];
    } else if (tid < NX + NQ + NP) {
        size_t t = tid - NX - NQ;
        int o = (int)(t / DIM), i = (int)(t % DIM);
        wprojT[t] = (_Float16)proj_w[(size_t)i * DIM + o];
    }
}

// ---------------- kernel B: QKV GEMM (4096 x 2304 x 768) ----------------
__global__ __launch_bounds__(128) void qkv_gemm(const _Float16* __restrict__ x_h,
                                                const _Float16* __restrict__ wT,
                                                const float* __restrict__ bias,
                                                _Float16* __restrict__ q_h,
                                                _Float16* __restrict__ k_h,
                                                _Float16* __restrict__ v_t) {
    const int lane = threadIdx.x & 31;
    const int wv   = threadIdx.x >> 5;
    const int bm   = blockIdx.x % (ROWS / 64);
    const int bn   = blockIdx.x / (ROWS / 64);
    const int m0   = bm * 64 + wv * 16;
    const int n0   = bn * 64;

    v8f acc[4] = {};
    for (int kk = 0; kk < DIM; kk += 32) {
        const _Float16* abase = x_h + (size_t)m0 * DIM + kk;
        v16h a = ld_fragA(abase, DIM, lane);
        if (kk + 160 < DIM) pf(abase + (size_t)(lane & 15) * DIM + 128);
#pragma unroll
        for (int t = 0; t < 4; ++t) {
            const _Float16* bbase = wT + (size_t)(n0 + t * 16) * DIM + kk;
            v16h b = ld_fragB(bbase, DIM, lane);
            if (kk + 160 < DIM) pf(bbase + (size_t)(lane & 15) * DIM + 128);
            acc[t] = wmma16(a, b, acc[t]);
        }
    }
#pragma unroll
    for (int t = 0; t < 4; ++t) {
#pragma unroll
        for (int r = 0; r < 8; ++r) {
            int row = m0 + r + ((lane >> 4) << 3);
            int col = n0 + t * 16 + (lane & 15);
            _Float16 hv = (_Float16)(acc[t][r] + bias[col]);
            int which = col / DIM;
            int c = col % DIM;
            int hh = c >> 6, d = c & 63;
            int b = row >> 10, tk = row & 1023;
            size_t head = (size_t)(b * HEADS + hh);
            if (which == 0)      q_h[(head * NTOK + tk) * DH + d] = hv;
            else if (which == 1) k_h[(head * NTOK + tk) * DH + d] = hv;
            else                 v_t[(head * DH + d) * NTOK + tk] = hv;
        }
    }
}

// ---------------- kernel C: fused attention per (head, 16-row block) ----------------
__global__ __launch_bounds__(256) void attn_kernel(const _Float16* __restrict__ q_h,
                                                   const _Float16* __restrict__ k_h,
                                                   const _Float16* __restrict__ v_t,
                                                   _Float16* __restrict__ out_h) {
    extern __shared__ char smem[];
    float*    sc = (float*)smem;                         // 16 x 1024 f32 scores (reused as partial buf)
    _Float16* ph = (_Float16*)(smem + 16 * NTOK * 4);    // 16 x 1024 f16 probs

    const int lane = threadIdx.x & 31;
    const int wv   = threadIdx.x >> 5;       // 0..7
    const int rb   = blockIdx.x & 63;        // row block within head
    const int head = blockIdx.x >> 6;        // 0..47 (= b*HEADS + h)
    const int m0   = rb * 16;

    const _Float16* qbase = q_h + ((size_t)head * NTOK + m0) * DH;
    const _Float16* kbase = k_h + (size_t)head * NTOK * DH;
    const _Float16* vbase = v_t + (size_t)head * DH * NTOK;

    // ---- step 1: scores = (q kT) * scale into LDS ----
    v16h a0 = ld_fragA(qbase, DH, lane);
    v16h a1 = ld_fragA(qbase + 32, DH, lane);
    const float scale = 0.125f;  // DH^-0.5
    for (int ct = wv * 8; ct < wv * 8 + 8; ++ct) {
        v16h b0 = ld_fragB(kbase + (size_t)(ct * 16) * DH, DH, lane);
        v16h b1 = ld_fragB(kbase + (size_t)(ct * 16) * DH + 32, DH, lane);
        v8f s = {};
        s = wmma16(a0, b0, s);
        s = wmma16(a1, b1, s);
#pragma unroll
        for (int r = 0; r < 8; ++r) {
            int row = r + ((lane >> 4) << 3);
            int col = ct * 16 + (lane & 15);
            sc[row * NTOK + col] = s[r] * scale;
        }
    }
    __syncthreads();

    // ---- step 2: exact top-512 threshold + softmax per row (2 rows/wave) ----
    for (int rr = 0; rr < 2; ++rr) {
        int row = wv * 2 + rr;
        float    vals[32];
        unsigned keys[32];
#pragma unroll
        for (int i = 0; i < 32; ++i) {
            vals[i] = sc[row * NTOK + lane + 32 * i];
            keys[i] = f2key(vals[i]);
        }
        // 32-bit radix binary search for the KSPARSE-th largest key
        unsigned P = 0;
        for (int bit = 31; bit >= 0; --bit) {
            unsigned cand = P | (1u << bit);
            int c = 0;
#pragma unroll
            for (int i = 0; i < 32; ++i) c += (keys[i] >= cand) ? 1 : 0;
#pragma unroll
            for (int off = 16; off; off >>= 1) c += __shfl_xor(c, off, 32);
            if (c >= KSPARSE) P = cand;
        }
        // row max (top-1 is always kept)
        float m = -3.0e38f;
#pragma unroll
        for (int i = 0; i < 32; ++i) m = fmaxf(m, vals[i]);
#pragma unroll
        for (int off = 16; off; off >>= 1) m = fmaxf(m, __shfl_xor(m, off, 32));
        // masked exp + sum
        float s = 0.f;
#pragma unroll
        for (int i = 0; i < 32; ++i) {
            float e = (keys[i] >= P) ? __expf(vals[i] - m) : 0.f;
            vals[i] = e;
            s += e;
        }
#pragma unroll
        for (int off = 16; off; off >>= 1) s += __shfl_xor(s, off, 32);
        float inv = 1.f / s;
#pragma unroll
        for (int i = 0; i < 32; ++i)
            ph[row * NTOK + lane + 32 * i] = (_Float16)(vals[i] * inv);
    }
    __syncthreads();

    // ---- step 3: out = P V (16 x 64, K = 1024), all 8 waves: K split in halves ----
    {
        const int tile = wv & 3;          // n-tile (d range tile*16..+15)
        const int half = wv >> 2;         // K half
        const int kb0  = half * (NTOK / 2);
        v8f acc = {};
        for (int kb = kb0; kb < kb0 + NTOK / 2; kb += 32) {
            v16h a = ld_fragA(ph + kb, NTOK, lane);   // probs from LDS
            const _Float16* bbase = vbase + (size_t)(tile * 16) * NTOK + kb;
            v16h b = ld_fragB(bbase, NTOK, lane);
            if (kb + 160 < kb0 + NTOK / 2) pf(bbase + (size_t)(lane & 15) * NTOK + 128);
            acc = wmma16(a, b, acc);
        }
        const int rl = ((lane >> 4) << 3);
        const int cl = lane & 15;
        if (wv >= 4) {
#pragma unroll
            for (int r = 0; r < 8; ++r)
                sc[(tile * 16 + r + rl) * 16 + cl] = acc[r];
        }
        __syncthreads();
        if (wv < 4) {
            int bb = head / HEADS, hh = head % HEADS;
#pragma unroll
            for (int r = 0; r < 8; ++r) {
                float v = acc[r] + sc[(tile * 16 + r + rl) * 16 + cl];
                int row = m0 + r + rl;
                int col = hh * DH + tile * 16 + cl;
                out_h[((size_t)bb * NTOK + row) * DIM + col] = (_Float16)v;
            }
        }
    }
}

// ---------------- kernel D1: multi-token mix  agg = multi @ mix_w + mix_b ----------------
__global__ void mix_kernel(const _Float16* __restrict__ out_h,
                           const float* __restrict__ mix_w,
                           const float* __restrict__ mix_b,
                           float* __restrict__ agg) {
    int tid = blockIdx.x * blockDim.x + threadIdx.x;
    if (tid >= BATCH * DIM) return;
    int b = tid / DIM, o = tid % DIM;
    const int tidx[4] = {0, 341, 682, 1023};
    float s = mix_b[o];
    for (int j = 0; j < 4; ++j) {
        const _Float16* rowp = out_h + ((size_t)b * NTOK + tidx[j]) * DIM;
        for (int c = 0; c < DIM; ++c)
            s += (float)rowp[c] * mix_w[((size_t)(j * DIM + c)) * DIM + o];
    }
    agg[tid] = s;
}

// ---------------- kernel D2: fold mix into per-batch bias: bias2 = 0.1*agg@proj_w + proj_b ----
__global__ void bias2_kernel(const float* __restrict__ agg,
                             const float* __restrict__ proj_w,
                             const float* __restrict__ proj_b,
                             float* __restrict__ bias2) {
    int tid = blockIdx.x * blockDim.x + threadIdx.x;
    if (tid >= BATCH * DIM) return;
    int b = tid / DIM, o2 = tid % DIM;
    float s = proj_b[o2];
    for (int o = 0; o < DIM; ++o)
        s += 0.1f * agg[b * DIM + o] * proj_w[(size_t)o * DIM + o2];
    bias2[tid] = s;
}

// ---------------- kernel E: final projection GEMM (4096 x 768 x 768) ----------------
__global__ __launch_bounds__(128) void proj_gemm(const _Float16* __restrict__ out_h,
                                                 const _Float16* __restrict__ wT,
                                                 const float* __restrict__ bias2,
                                                 float* __restrict__ y) {
    const int lane = threadIdx.x & 31;
    const int wv   = threadIdx.x >> 5;
    const int bm   = blockIdx.x % (ROWS / 64);
    const int bn   = blockIdx.x / (ROWS / 64);
    const int m0   = bm * 64 + wv * 16;
    const int n0   = bn * 64;

    v8f acc[4] = {};
    for (int kk = 0; kk < DIM; kk += 32) {
        const _Float16* abase = out_h + (size_t)m0 * DIM + kk;
        v16h a = ld_fragA(abase, DIM, lane);
        if (kk + 160 < DIM) pf(abase + (size_t)(lane & 15) * DIM + 128);
#pragma unroll
        for (int t = 0; t < 4; ++t) {
            const _Float16* bbase = wT + (size_t)(n0 + t * 16) * DIM + kk;
            v16h b = ld_fragB(bbase, DIM, lane);
            if (kk + 160 < DIM) pf(bbase + (size_t)(lane & 15) * DIM + 128);
            acc[t] = wmma16(a, b, acc[t]);
        }
    }
#pragma unroll
    for (int t = 0; t < 4; ++t) {
#pragma unroll
        for (int r = 0; r < 8; ++r) {
            int row = m0 + r + ((lane >> 4) << 3);
            int col = n0 + t * 16 + (lane & 15);
            int b = row >> 10;
            y[(size_t)row * DIM + col] = acc[t][r] + bias2[b * DIM + col];
        }
    }
}

// ---------------- launcher ----------------
extern "C" void kernel_launch(void* const* d_in, const int* in_sizes, int n_in,
                              void* d_out, int out_size, void* d_ws, size_t ws_size,
                              hipStream_t stream) {
    const float* x      = (const float*)d_in[0];
    const float* qkv_w  = (const float*)d_in[1];
    const float* qkv_b  = (const float*)d_in[2];
    const float* proj_w = (const float*)d_in[3];
    const float* proj_b = (const float*)d_in[4];
    const float* mix_w  = (const float*)d_in[5];
    const float* mix_b  = (const float*)d_in[6];
    float* y = (float*)d_out;

    // workspace carve-up (bytes, all 256-aligned sizes)
    char* ws = (char*)d_ws;
    const size_t SZ_XH    = (size_t)ROWS * DIM * 2;        // 6291456
    const size_t SZ_WQKVT = (size_t)QKV_N * DIM * 2;       // 3538944
    const size_t SZ_WPROJ = (size_t)DIM * DIM * 2;         // 1179648
    const size_t SZ_QKV1  = (size_t)BATCH * HEADS * NTOK * DH * 2; // 6291456
    const size_t SZ_OUTH  = (size_t)ROWS * DIM * 2;        // 6291456
    const size_t SZ_AGG   = (size_t)BATCH * DIM * 4;       // 12288

    size_t off = 0;
    _Float16* x_h    = (_Float16*)(ws + off); off += SZ_XH;
    _Float16* wqkvT  = (_Float16*)(ws + off); off += SZ_WQKVT;
    _Float16* wprojT = (_Float16*)(ws + off); off += SZ_WPROJ;
    _Float16* q_h    = (_Float16*)(ws + off); off += SZ_QKV1;
    _Float16* k_h    = (_Float16*)(ws + off); off += SZ_QKV1;
    _Float16* v_t    = (_Float16*)(ws + off); off += SZ_QKV1;
    _Float16* out_h  = (_Float16*)(ws + off); off += SZ_OUTH;
    float*    agg    = (float*)(ws + off);    off += SZ_AGG;
    float*    bias2  = (float*)(ws + off);    off += SZ_AGG;

    // A: precision/layout prep
    {
        size_t total = (size_t)ROWS * DIM + (size_t)DIM * QKV_N + (size_t)DIM * DIM;
        int grid = (int)((total + 255) / 256);
        prep_kernel<<<grid, 256, 0, stream>>>(x, qkv_w, proj_w, x_h, wqkvT, wprojT);
    }
    // B: QKV GEMM
    qkv_gemm<<<(ROWS / 64) * (QKV_N / 64), 128, 0, stream>>>(x_h, wqkvT, qkv_b, q_h, k_h, v_t);
    // C: fused sparse attention
    {
        int grid = BATCH * HEADS * (NTOK / 16);     // 3072
        size_t shmem = (size_t)16 * NTOK * 4 + (size_t)16 * NTOK * 2; // 96 KB
        attn_kernel<<<grid, 256, shmem, stream>>>(q_h, k_h, v_t, out_h);
    }
    // D1/D2: mixing folded into per-batch bias
    mix_kernel<<<(BATCH * DIM + 255) / 256, 256, 0, stream>>>(out_h, mix_w, mix_b, agg);
    bias2_kernel<<<(BATCH * DIM + 255) / 256, 256, 0, stream>>>(agg, proj_w, proj_b, bias2);
    // E: final projection
    proj_gemm<<<(ROWS / 64) * (DIM / 64), 128, 0, stream>>>(out_h, wprojT, bias2, y);
}
